// CausalSelfAttention_26018911879793
// MI455X (gfx1250) — matrix operand order repaired and checked by reading, not compiled, not run on hardware
//
#include <hip/hip_runtime.h>

typedef __bf16 bf16;
typedef __attribute__((ext_vector_type(16))) __bf16 v16bf;
typedef __attribute__((ext_vector_type(8)))  __bf16 v8bf;
typedef __attribute__((ext_vector_type(8)))  float  v8f;

#define B_   2
#define T_   2048
#define C_   2048
#define H_   16
#define KVH_ 4
#define HD_  128

__device__ __forceinline__ v16bf cat8(v8bf lo, v8bf hi) {
  return __builtin_shufflevector(lo, hi, 0,1,2,3,4,5,6,7,8,9,10,11,12,13,14,15);
}

// ---------------------------------------------------------------------------
// x -> bf16 (row-major, used as GEMM A operand)
// ---------------------------------------------------------------------------
__global__ void cvt_f32_bf16(const float* __restrict__ in, bf16* __restrict__ out, size_t n) {
  size_t i = (size_t)blockIdx.x * blockDim.x + threadIdx.x;
  size_t stride = (size_t)gridDim.x * blockDim.x;
  for (; i < n; i += stride) out[i] = (bf16)in[i];
}

// ---------------------------------------------------------------------------
// Tiled transpose+convert: in[R][Cc] f32 -> out[Cc][R] bf16 (coalesced both ways)
// grid (Cc/32, R/32), block (32,8)
// ---------------------------------------------------------------------------
__global__ void __launch_bounds__(256) transpose_cvt(
    const float* __restrict__ in, bf16* __restrict__ out, int R, int Cc) {
  __shared__ float tile[32][33];
  const int c0 = blockIdx.x * 32, r0 = blockIdx.y * 32;
  const int tx = threadIdx.x, ty = threadIdx.y;
#pragma unroll
  for (int i = 0; i < 4; i++)
    tile[ty + i * 8][tx] = in[(size_t)(r0 + ty + i * 8) * Cc + c0 + tx];
  __syncthreads();
#pragma unroll
  for (int i = 0; i < 4; i++)
    out[(size_t)(c0 + ty + i * 8) * R + r0 + tx] = (bf16)tile[tx][ty + i * 8];
}

// ---------------------------------------------------------------------------
// Pack wq|wk|wv into one logical [2048 x 3072] matrix, transposed to
// out[3072][2048] bf16.  grid (3072/32, 2048/32), block (32,8)
// ---------------------------------------------------------------------------
__global__ void __launch_bounds__(256) pack_wqkv_T(
    const float* __restrict__ wq, const float* __restrict__ wk,
    const float* __restrict__ wv, bf16* __restrict__ out) {
  __shared__ float tile[32][33];
  const int c0 = blockIdx.x * 32, r0 = blockIdx.y * 32;
  const int tx = threadIdx.x, ty = threadIdx.y;
#pragma unroll
  for (int i = 0; i < 4; i++) {
    const int r = r0 + ty + i * 8, c = c0 + tx;
    float v;
    if (c < 2048)      v = wq[(size_t)r * 2048 + c];
    else if (c < 2560) v = wk[(size_t)r * 512 + (c - 2048)];
    else               v = wv[(size_t)r * 512 + (c - 2560)];
    tile[ty + i * 8][tx] = v;
  }
  __syncthreads();
#pragma unroll
  for (int i = 0; i < 4; i++)
    out[(size_t)(c0 + ty + i * 8) * 2048 + r0 + tx] = (bf16)tile[tx][ty + i * 8];
}

// ---------------------------------------------------------------------------
// WMMA GEMM with pre-transposed B:
//   C[M,N](f32) = A[M,K](bf16 rowmajor) * Bt[N,K](bf16 rowmajor)^T
// No LDS: every fragment is a contiguous per-lane global load (L2-resident B).
// Block = 256 (8 waves, 4x2), block tile 128x128, wave tile 32x64, K step 32.
// ---------------------------------------------------------------------------
__global__ void __launch_bounds__(256) gemm_bt_bf16_f32(
    const bf16* __restrict__ A, const bf16* __restrict__ Bt,
    float* __restrict__ C, int M, int N, int K) {
  const int tid  = threadIdx.x;
  const int wave = tid >> 5;
  const int lane = tid & 31;
  const int hh   = lane >> 4;     // K-half selector
  const int nn   = lane & 15;     // M row (A frag) / N col (B,C frags)
  const int wm   = wave & 3;      // wave M quadrant (rows wm*32)
  const int wn   = wave >> 2;     // wave N half     (cols wn*64)
  const int m0   = blockIdx.y * 128;
  const int n0   = blockIdx.x * 128;

  const v8f vz = {0.f,0.f,0.f,0.f,0.f,0.f,0.f,0.f};
  v8f acc[2][4];
#pragma unroll
  for (int i = 0; i < 2; i++)
#pragma unroll
    for (int j = 0; j < 4; j++) acc[i][j] = vz;

  const bf16* Arow0 = A  + (size_t)(m0 + wm * 32 + nn) * K;        // +i*16 rows
  const bf16* Brow0 = Bt + (size_t)(n0 + wn * 64 + nn) * K;        // +j*16 rows

#pragma unroll 2
  for (int k0 = 0; k0 < K; k0 += 32) {
    v16bf af[2];
#pragma unroll
    for (int i = 0; i < 2; i++) {
      const bf16* p = Arow0 + (size_t)i * 16 * K + k0;
      af[i] = cat8(*(const v8bf*)(p + hh * 16), *(const v8bf*)(p + hh * 16 + 8));
    }
#pragma unroll
    for (int j = 0; j < 4; j++) {
      const v16bf bfv = *(const v16bf*)(Brow0 + (size_t)j * 16 * K + k0 + hh * 16);
#pragma unroll
      for (int i = 0; i < 2; i++)
        acc[i][j] = __builtin_amdgcn_wmma_f32_16x16x32_bf16(
            false, af[i], false, bfv, (short)0, acc[i][j], false, false);
    }
  }

#pragma unroll
  for (int i = 0; i < 2; i++)
#pragma unroll
    for (int j = 0; j < 4; j++) {
      float* Cp = C + (size_t)(m0 + wm * 32 + i * 16 + 8 * hh) * N + n0 + wn * 64 + j * 16 + nn;
#pragma unroll
      for (int r = 0; r < 8; r++) Cp[(size_t)r * N] = acc[i][j][r];
    }
}

// ---------------------------------------------------------------------------
// RoPE + RMSNorm on Q,K; V transpose.  qkv f32 [B*T, 3072] ->
//   q bf16 [B,H,T,128], k bf16 [B,KVH,T,128], vT bf16 [B,KVH,128,T]
// grid = (B*T, H + 2*KVH), block = 128
// ---------------------------------------------------------------------------
__global__ void __launch_bounds__(128) rope_norm_kernel(
    const float* __restrict__ qkv, const float* __restrict__ cosb,
    const float* __restrict__ sinb, const float* __restrict__ qkw,
    bf16* __restrict__ qo, bf16* __restrict__ ko, bf16* __restrict__ vto) {
  const int row = blockIdx.x;           // b*T + t
  const int b = row / T_, t = row % T_;
  const int hy = blockIdx.y;            // 0..23
  const int d = threadIdx.x;
  const float* xr = qkv + (size_t)row * 3072;

  if (hy >= H_ + KVH_) {                // V heads: no rope/norm, transpose store
    const int vh = hy - (H_ + KVH_);
    float val = xr[2560 + vh * HD_ + d];
    vto[((size_t)(b * KVH_ + vh) * HD_ + d) * T_ + t] = (bf16)val;
    return;
  }
  const int col = (hy < H_) ? hy * HD_ : 2048 + (hy - H_) * HD_;
  const float x1 = xr[col + d];
  float val;
  if (d < 64) {
    float x2 = xr[col + d + 64];
    val = x1 * cosb[t * 64 + d] - x2 * sinb[t * 64 + d];
  } else {
    float x2 = xr[col + d - 64];
    val = x1 * cosb[t * 64 + d - 64] + x2 * sinb[t * 64 + d - 64];
  }
  float ss = val * val;
#pragma unroll
  for (int m = 1; m < 32; m <<= 1) ss += __shfl_xor(ss, m, 32);
  __shared__ float part[4];
  if ((threadIdx.x & 31) == 0) part[threadIdx.x >> 5] = ss;
  __syncthreads();
  const float tot = part[0] + part[1] + part[2] + part[3];
  const float rs = rsqrtf(tot * (1.0f / 128.0f) + 1e-6f);
  const float outv = val * rs * qkw[d];
  if (hy < H_)
    qo[((size_t)(b * H_ + hy) * T_ + t) * HD_ + d] = (bf16)outv;
  else
    ko[((size_t)(b * KVH_ + (hy - H_)) * T_ + t) * HD_ + d] = (bf16)outv;
}

// ---------------------------------------------------------------------------
// Flash attention (causal, GQA).  grid = (T/128, H, B), block = 256 (8 waves).
// Each wave owns 16 q rows; k-tiles of 32 with online softmax.
// All WMMA operands are contiguous per-lane loads (K rowmajor, V transposed).
// ---------------------------------------------------------------------------
__global__ void __launch_bounds__(256) flash_attn_kernel(
    const bf16* __restrict__ Q, const bf16* __restrict__ Kb,
    const bf16* __restrict__ Vt, bf16* __restrict__ Y) {
  __shared__ bf16 pbuf[8][16 * 40];     // per-wave 16x32 P tile, row stride 40

  const int tid = threadIdx.x, wave = tid >> 5, lane = tid & 31;
  const int hh = lane >> 4, nn = lane & 15;
  const int qt = blockIdx.x, h = blockIdx.y, b = blockIdx.z;
  const int kvh = h >> 2;

  const bf16* Qp = Q  + (size_t)(b * H_ + h) * T_ * HD_;
  const bf16* Kp = Kb + (size_t)(b * KVH_ + kvh) * T_ * HD_;
  const bf16* Vp = Vt + (size_t)(b * KVH_ + kvh) * HD_ * T_;
  const int qbase = qt * 128 + wave * 16;

  // preload Q A-fragments (16 rows x 128, four K=32 chunks)
  v16bf aq[4];
#pragma unroll
  for (int kk = 0; kk < 4; kk++) {
    const bf16* qr = Qp + (size_t)(qbase + nn) * HD_ + kk * 32;
    aq[kk] = cat8(*(const v8bf*)(qr + hh * 16), *(const v8bf*)(qr + hh * 16 + 8));
  }

  const v8f vz = {0.f,0.f,0.f,0.f,0.f,0.f,0.f,0.f};
  v8f o[8];
#pragma unroll
  for (int i = 0; i < 8; i++) o[i] = vz;
  float mrow[8], lrow[8];
#pragma unroll
  for (int r = 0; r < 8; r++) { mrow[r] = -1e30f; lrow[r] = 0.f; }
  const float scale = 0.08838834764831845f;  // 1/sqrt(128)

  const int ktend = (qbase + 15) >> 5;
  for (int kt = 0; kt <= ktend; kt++) {
    const int kc0 = kt * 32;
    v8f s0 = vz, s1 = vz;
#pragma unroll
    for (int kk = 0; kk < 4; kk++) {
      const v16bf bk0 = *(const v16bf*)(Kp + (size_t)(kc0 + nn) * HD_ + kk * 32 + hh * 16);
      const v16bf bk1 = *(const v16bf*)(Kp + (size_t)(kc0 + 16 + nn) * HD_ + kk * 32 + hh * 16);
      s0 = __builtin_amdgcn_wmma_f32_16x16x32_bf16(false, aq[kk], false, bk0, (short)0, s0, false, false);
      s1 = __builtin_amdgcn_wmma_f32_16x16x32_bf16(false, aq[kk], false, bk1, (short)0, s1, false, false);
    }
    float alpha[8];
#pragma unroll
    for (int r = 0; r < 8; r++) {
      const int qrow = qbase + r + 8 * hh;
      float v0 = s0[r] * scale; if (kc0 + nn > qrow)      v0 = -1e30f;
      float v1 = s1[r] * scale; if (kc0 + 16 + nn > qrow) v1 = -1e30f;
      float tm = fmaxf(v0, v1);
      tm = fmaxf(tm, __shfl_xor(tm, 1, 32));
      tm = fmaxf(tm, __shfl_xor(tm, 2, 32));
      tm = fmaxf(tm, __shfl_xor(tm, 4, 32));
      tm = fmaxf(tm, __shfl_xor(tm, 8, 32));
      const float mnew = fmaxf(mrow[r], tm);
      const float a = __expf(mrow[r] - mnew);
      const float p0 = __expf(v0 - mnew), p1 = __expf(v1 - mnew);
      float rsum = p0 + p1;
      rsum += __shfl_xor(rsum, 1, 32);
      rsum += __shfl_xor(rsum, 2, 32);
      rsum += __shfl_xor(rsum, 4, 32);
      rsum += __shfl_xor(rsum, 8, 32);
      lrow[r] = lrow[r] * a + rsum;
      mrow[r] = mnew;
      alpha[r] = a;
      pbuf[wave][(r + 8 * hh) * 40 + nn]      = (bf16)p0;   // D-layout -> LDS
      pbuf[wave][(r + 8 * hh) * 40 + 16 + nn] = (bf16)p1;
    }
    // reload P as A-fragment (LDS ops in-order within a wave)
    const v16bf ap = cat8(*(const v8bf*)(&pbuf[wave][nn * 40 + hh * 16]),
                          *(const v8bf*)(&pbuf[wave][nn * 40 + hh * 16 + 8]));
#pragma unroll
    for (int ns = 0; ns < 8; ns++) {
#pragma unroll
      for (int r = 0; r < 8; r++) o[ns][r] *= alpha[r];
      const v16bf bv = *(const v16bf*)(Vp + (size_t)(ns * 16 + nn) * T_ + kc0 + hh * 16);
      o[ns] = __builtin_amdgcn_wmma_f32_16x16x32_bf16(false, ap, false, bv, (short)0, o[ns], false, false);
    }
  }

#pragma unroll
  for (int ns = 0; ns < 8; ns++)
#pragma unroll
    for (int r = 0; r < 8; r++) {
      const int qrow = qbase + r + 8 * hh;
      const float val = o[ns][r] / lrow[r];
      Y[(size_t)(b * T_ + qrow) * C_ + h * HD_ + ns * 16 + nn] = (bf16)val;
    }
}

// ---------------------------------------------------------------------------
extern "C" void kernel_launch(void* const* d_in, const int* in_sizes, int n_in,
                              void* d_out, int out_size, void* d_ws, size_t ws_size,
                              hipStream_t stream) {
  const float* x    = (const float*)d_in[0];
  const float* cosb = (const float*)d_in[1];
  const float* sinb = (const float*)d_in[2];
  const float* wq   = (const float*)d_in[3];
  const float* wk   = (const float*)d_in[4];
  const float* wv   = (const float*)d_in[5];
  const float* wo   = (const float*)d_in[6];
  const float* qkw  = (const float*)d_in[7];
  float* out = (float*)d_out;
  char* ws = (char*)d_ws;

  bf16*  xbf   = (bf16*)(ws + 0);            // 16.8 MB  [4096,2048]
  bf16*  wqkvT = (bf16*)(ws + 16777216);     // 12.6 MB  [3072,2048] (B^T)
  bf16*  woT   = (bf16*)(ws + 29360128);     //  8.4 MB  [2048,2048] (B^T)
  float* qkv   = (float*)(ws + 37748736);    // 50.3 MB  [4096,3072]
  bf16*  qb    = (bf16*)(ws + 88080384);     // 16.8 MB  [B,H,T,128]
  bf16*  kb    = (bf16*)(ws + 104857600);    //  4.2 MB  [B,KVH,T,128]
  bf16*  vtb   = (bf16*)(ws + 109051904);    //  4.2 MB  [B,KVH,128,T]
  bf16*  yb    = (bf16*)(ws + 113246208);    // 16.8 MB  [4096,2048]

  cvt_f32_bf16<<<2048, 256, 0, stream>>>(x, xbf, (size_t)B_ * T_ * C_);
  pack_wqkv_T  <<<dim3(3072 / 32, 2048 / 32), dim3(32, 8), 0, stream>>>(wq, wk, wv, wqkvT);
  transpose_cvt<<<dim3(2048 / 32, 2048 / 32), dim3(32, 8), 0, stream>>>(wo, woT, 2048, 2048);

  // QKV projection: [4096,2048] x [2048,3072] -> [4096,3072] f32
  gemm_bt_bf16_f32<<<dim3(3072 / 128, 4096 / 128), 256, 0, stream>>>(
      xbf, wqkvT, qkv, 4096, 3072, 2048);

  rope_norm_kernel<<<dim3(B_ * T_, H_ + 2 * KVH_), 128, 0, stream>>>(
      qkv, cosb, sinb, qkw, qb, kb, vtb);

  flash_attn_kernel<<<dim3(T_ / 128, H_, B_), 256, 0, stream>>>(qb, kb, vtb, yb);

  // Output projection: [4096,2048] x [2048,2048] -> [4096,2048] f32
  gemm_bt_bf16_f32<<<dim3(2048 / 128, 4096 / 128), 256, 0, stream>>>(
      yb, woT, out, 4096, 2048, 2048);
}
